// MultiHeadAttention_66211215835323
// MI455X (gfx1250) — matrix operand order repaired and checked by reading, not compiled, not run on hardware
//
#include <hip/hip_runtime.h>
#include <hip/hip_bf16.h>
#include <stdint.h>

typedef __attribute__((ext_vector_type(16))) _Float16 v16h;
typedef __attribute__((ext_vector_type(8)))  float    v8f;
typedef _Float16 half_t;

static __device__ inline v8f wmma_f16(v16h a, v16h b, v8f c) {
  // D = A(16x32 f16) * B(32x16 f16) + C(16x16 f32)
  return __builtin_amdgcn_wmma_f32_16x16x32_f16(false, a, false, b, (short)0, c, false, false);
}

// ---- Fragment loaders -------------------------------------------------------
// A fragment (16x32 f16): lane L holds row m = L%16; g = L/16 selects K-halves.
// VGPR v(0..3): k = g*8 + v*2 + {0,1}; VGPR v(4..7): k = 16 + g*8 + (v-4)*2 + {0,1}.
// Each half coalesces into one global_load_b128.
static __device__ inline v16h load_frag_a(const half_t* __restrict__ A, int lda,
                                          int row, int kBase, int lane) {
  const int m = lane & 15, g = lane >> 4;
  const half_t* p = A + (size_t)(row + m) * lda + kBase;
  union { v16h v; unsigned u[8]; } f;
#pragma unroll
  for (int vr = 0; vr < 8; ++vr) {
    const int kk = (vr < 4) ? (g * 8 + vr * 2) : (16 + g * 8 + (vr - 4) * 2);
    f.u[vr] = *(const unsigned*)(p + kk);
  }
  return f.v;
}

// A fragment where real K = 16, zero-padded to 32 (upper 4 VGPRs zero).
static __device__ inline v16h load_frag_a_k16(const half_t* __restrict__ A, int lda,
                                              int row, int lane) {
  const int m = lane & 15, g = lane >> 4;
  const half_t* p = A + (size_t)(row + m) * lda;
  union { v16h v; unsigned u[8]; } f;
#pragma unroll
  for (int vr = 0; vr < 4; ++vr) f.u[vr] = *(const unsigned*)(p + g * 8 + vr * 2);
#pragma unroll
  for (int vr = 4; vr < 8; ++vr) f.u[vr] = 0u;
  return f.v;
}

// B fragment (32x16 f16), operand stored transposed: Bt[col][k] row-major.
// Lane L holds column n = L%16; same k order as A: v<4: k = g*8 + v*2, v>=4: k = 16 + g*8 + (v-4)*2.
static __device__ inline v16h load_frag_b(const half_t* __restrict__ Bt, int ldb,
                                          int col, int kBase, int lane) {
  const int n = lane & 15, g = lane >> 4;
  const half_t* p = Bt + (size_t)(col + n) * ldb + kBase;
  union { v16h v; unsigned u[8]; } f;
#pragma unroll
  for (int vr = 0; vr < 8; ++vr) f.u[vr] = *(const unsigned*)(p + ((vr < 4) ? (g * 8 + vr * 2) : (16 + g * 8 + (vr - 4) * 2)));
  return f.v;
}

// B fragment with real K = 16, zero-padded to 32 (upper 4 VGPRs zero), same k order as A.
static __device__ inline v16h load_frag_b_k16(const half_t* __restrict__ Bt, int ldb,
                                              int col, int lane) {
  const int n = lane & 15, g = lane >> 4;
  union { v16h v; unsigned u[8]; } f;
  {
    const half_t* p = Bt + (size_t)(col + n) * ldb;
#pragma unroll
    for (int vr = 0; vr < 4; ++vr) f.u[vr] = *(const unsigned*)(p + g * 8 + vr * 2);
#pragma unroll
    for (int vr = 4; vr < 8; ++vr) f.u[vr] = 0u;
  }
  return f.v;
}

// A fragment from a private [16][32] f16 LDS tile.
static __device__ inline v16h load_frag_a_lds(const half_t* p0, int lane) {
  const int m = lane & 15, g = lane >> 4;
  const half_t* p = p0 + m * 32;
  union { v16h v; unsigned u[8]; } f;
#pragma unroll
  for (int vr = 0; vr < 8; ++vr) {
    const int kk = (vr < 4) ? (g * 8 + vr * 2) : (16 + g * 8 + (vr - 4) * 2);
    f.u[vr] = *(const unsigned*)(p + kk);
  }
  return f.v;
}

// ---- Stage 0: fp32 -> f16 conversion ---------------------------------------
__global__ void f32_to_f16_kernel(const float* __restrict__ src,
                                  half_t* __restrict__ dst, int n) {
  int i = blockIdx.x * blockDim.x + threadIdx.x;
  if (i < n) dst[i] = (half_t)src[i];
}

// ---- Stage 1/4: generic f16 WMMA GEMM  C[M,N] = A[M,K] * Bt[N,K]^T ---------
// 32x64 tile per wave: 2 A-frags reuse each B-frag twice (8 WMMA / 12 b128).
// MODE 0: f16 out row-major      MODE 1: f16 out, column index swapped (kd*16+h)
// MODE 2: f16 out transposed [N][M]   MODE 3: f32 out row-major
template <int MODE>
__global__ void __launch_bounds__(256)
gemm_f16_kernel(const half_t* __restrict__ A, const half_t* __restrict__ Bt,
                void* __restrict__ Cout, int M, int N, int K, int ldc) {
  (void)M; (void)N;
  const int lane = threadIdx.x & 31;
  const int wave = threadIdx.x >> 5;
  const int row = (blockIdx.x * 8 + wave) * 32; // 32 rows per wave
  const int colBase = blockIdx.y * 64;          // 64 cols per wave
  const v8f zero = {0.f, 0.f, 0.f, 0.f, 0.f, 0.f, 0.f, 0.f};
  v8f acc[2][4] = {{zero, zero, zero, zero}, {zero, zero, zero, zero}};

  for (int k = 0; k < K; k += 32) {
    v16h a0 = load_frag_a(A, K, row, k, lane);
    v16h a1 = load_frag_a(A, K, row + 16, k, lane);
#pragma unroll
    for (int t = 0; t < 4; ++t) {
      v16h b = load_frag_b(Bt, K, colBase + t * 16, k, lane);
      acc[0][t] = wmma_f16(a0, b, acc[0][t]);
      acc[1][t] = wmma_f16(a1, b, acc[1][t]);
    }
  }
  const int n16 = lane & 15, g = lane >> 4;
#pragma unroll
  for (int s = 0; s < 2; ++s) {
#pragma unroll
    for (int t = 0; t < 4; ++t) {
#pragma unroll
      for (int r = 0; r < 8; ++r) {
        const int m = row + s * 16 + r + 8 * g; // C layout: m = r + 8*(lane/16)
        const int c = colBase + t * 16 + n16;   //            n = lane%16
        const float v = acc[s][t][r];
        if (MODE == 0) {
          ((half_t*)Cout)[(size_t)m * ldc + c] = (half_t)v;
        } else if (MODE == 1) {
          const int c2 = (c & 15) * 16 + (c >> 4); // (h*16+kd) -> (kd*16+h)
          ((half_t*)Cout)[(size_t)m * ldc + c2] = (half_t)v;
        } else if (MODE == 2) {
          ((half_t*)Cout)[(size_t)c * ldc + m] = (half_t)v;
        } else {
          ((float*)Cout)[(size_t)m * ldc + c] = v;
        }
      }
    }
  }
}

// ---- Stage 2: per-column (over n) online softmax stats ----------------------
// sim[b,n,h,m] = sum_j qswap[n][h*16+j] * kh[m][h*16+j]; softmax over n.
__global__ void __launch_bounds__(256)
colstats_kernel(const half_t* __restrict__ qswap, const half_t* __restrict__ kh,
                float* __restrict__ mx, float* __restrict__ cs) {
  const int lane = threadIdx.x & 31;
  const int wave = threadIdx.x >> 5;
  const int w = blockIdx.x * 8 + wave;        // 4096 waves total
  const int mt16 = (w & 127) * 16;
  const int h = (w >> 7) & 15;
  const int b = w >> 11;
  const float scale = 0.25f;                  // 1/sqrt(16)
  const int n16 = lane & 15;
  const v8f zero = {0.f, 0.f, 0.f, 0.f, 0.f, 0.f, 0.f, 0.f};

  // Key tile for this wave's 16 m-columns is loop-invariant.
  v16h bk = load_frag_b_k16(kh + h * 16, 256, b * 2048 + mt16, lane);
  const half_t* qbase = qswap + h * 16;

  float runMax = -3.4e38f, runSum = 0.f;
  for (int nt = 0; nt < 2048; nt += 16) {
    v16h aq = load_frag_a_k16(qbase, 256, b * 2048 + nt, lane);
    if (nt + 16 < 2048) // stream-ahead: global_prefetch next Q tile
      __builtin_prefetch(qbase + (size_t)(b * 2048 + nt + 16 + n16) * 256, 0, 3);
    v8f s = wmma_f16(aq, bk, zero);
    float tmax = -3.4e38f;
#pragma unroll
    for (int r = 0; r < 8; ++r) tmax = fmaxf(tmax, s[r] * scale);
    tmax = fmaxf(tmax, __shfl_xor(tmax, 16, 32)); // partner lane shares column m
    const float newMax = fmaxf(runMax, tmax);
    const float corr = __expf(runMax - newMax);
    float tsum = 0.f;
#pragma unroll
    for (int r = 0; r < 8; ++r) tsum += __expf(s[r] * scale - newMax);
    runSum = runSum * corr + tsum;
    runMax = newMax;
  }
  runSum += __shfl_xor(runSum, 16, 32);       // both lanes used same max sequence
  if (lane < 16) {
    const int idx = (b * 16 + h) * 2048 + mt16 + n16;
    mx[idx] = runMax;
    cs[idx] = runSum;
  }
}

// ---- Stage 3: fused attn = exp(S*scale - mx)/c, O = attn * V ----------------
__global__ void __launch_bounds__(256)
attn_o_kernel(const half_t* __restrict__ qswap, const half_t* __restrict__ kh,
              const half_t* __restrict__ vt, const float* __restrict__ mx,
              const float* __restrict__ cs, half_t* __restrict__ oh) {
  __shared__ half_t etile[8][16][32];         // private 16x32 E-tile per wave
  const int lane = threadIdx.x & 31;
  const int wave = threadIdx.x >> 5;
  const int w = blockIdx.x * 8 + wave;        // 4096 waves total
  const int nt = (w & 127) * 16;
  const int h = (w >> 7) & 15;
  const int b = w >> 11;
  const float scale = 0.25f;
  const int n16 = lane & 15, g = lane >> 4;
  const v8f zero = {0.f, 0.f, 0.f, 0.f, 0.f, 0.f, 0.f, 0.f};
  v8f acc[4] = {zero, zero, zero, zero};

  v16h aq = load_frag_a_k16(qswap + h * 16, 256, b * 2048 + nt, lane); // invariant
  const half_t* kbase = kh + h * 16;
  const float* mxp = mx + (size_t)(b * 16 + h) * 2048;
  const float* cpp = cs + (size_t)(b * 16 + h) * 2048;
  half_t* myTile = &etile[wave][0][0];

  for (int mt = 0; mt < 2048; mt += 32) {
    if (mt + 32 < 2048) // stream-ahead: global_prefetch next K tile
      __builtin_prefetch(kbase + (size_t)(b * 2048 + mt + 32 + n16) * 256, 0, 3);
#pragma unroll
    for (int hf = 0; hf < 2; ++hf) {
      const int mcol = mt + hf * 16;
      v16h bk = load_frag_b_k16(kbase, 256, b * 2048 + mcol, lane);
      v8f s = wmma_f16(aq, bk, zero);
      const int m = mcol + n16;               // this lane's column
      const float mval = mxp[m];
      const float inv = 1.0f / cpp[m];
#pragma unroll
      for (int r = 0; r < 8; ++r) {
        const float e = __expf(s[r] * scale - mval) * inv;
        myTile[(r + 8 * g) * 32 + hf * 16 + n16] = (half_t)e; // row n, col m-local
      }
    }
    __syncthreads(); // fence: E-tile writes (f16) before type-punned reads
    v16h ea = load_frag_a_lds(myTile, lane);  // repack C-layout -> A-layout
#pragma unroll
    for (int t = 0; t < 4; ++t) {
      v16h bv = load_frag_b(vt, 4096, h * 64 + t * 16, b * 2048 + mt, lane);
      acc[t] = wmma_f16(ea, bv, acc[t]);
    }
    __syncthreads(); // fence: reads complete before next iteration overwrites
  }
#pragma unroll
  for (int t = 0; t < 4; ++t)
#pragma unroll
    for (int r = 0; r < 8; ++r)
      oh[(size_t)(b * 2048 + nt + r + 8 * g) * 1024 + h * 64 + t * 16 + n16] =
          (half_t)acc[t][r];
}

// ---- Host-side orchestration ------------------------------------------------
extern "C" void kernel_launch(void* const* d_in, const int* in_sizes, int n_in,
                              void* d_out, int out_size, void* d_ws, size_t ws_size,
                              hipStream_t stream) {
  (void)in_sizes; (void)n_in; (void)out_size; (void)ws_size;
  const float* x  = (const float*)d_in[0]; // [2,2048,1024]
  const float* qp = (const float*)d_in[1]; // [16,16,1024]
  const float* kp = (const float*)d_in[2]; // [16,16,1024]
  const float* vp = (const float*)d_in[3]; // [16,64,1024]
  const float* op = (const float*)d_in[4]; // [1024,16,64]

  half_t* ws  = (half_t*)d_ws;
  half_t* xh  = ws;                  // [4096][1024]
  half_t* wq  = xh + 4194304;        // [256][1024]  (== query_proj layout)
  half_t* wk  = wq + 262144;         // [256][1024]
  half_t* wv  = wk + 262144;         // [1024][1024]
  half_t* wo  = wv + 1048576;        // [1024][1024] (== output_proj layout)
  half_t* qsw = wo + 1048576;        // [4096][256]  column-swapped Q
  half_t* khb = qsw + 1048576;       // [4096][256]
  half_t* vt  = khb + 1048576;       // [1024][4096] V transposed
  half_t* oh  = vt + 4194304;        // [4096][1024]
  float*  mxs = (float*)(oh + 4194304); // [2*16*2048]
  float*  css = mxs + 65536;            // [2*16*2048]

  f32_to_f16_kernel<<<(4194304 + 255) / 256, 256, 0, stream>>>(x,  xh, 4194304);
  f32_to_f16_kernel<<<(262144  + 255) / 256, 256, 0, stream>>>(qp, wq, 262144);
  f32_to_f16_kernel<<<(262144  + 255) / 256, 256, 0, stream>>>(kp, wk, 262144);
  f32_to_f16_kernel<<<(1048576 + 255) / 256, 256, 0, stream>>>(vp, wv, 1048576);
  f32_to_f16_kernel<<<(1048576 + 255) / 256, 256, 0, stream>>>(op, wo, 1048576);

  dim3 blk(256);
  // Q = X*Wq^T (cols swapped to kd*16+h), K = X*Wk^T, V = X*Wv^T (transposed out)
  gemm_f16_kernel<1><<<dim3(16, 4),  blk, 0, stream>>>(xh, wq, qsw, 4096, 256,  1024, 256);
  gemm_f16_kernel<0><<<dim3(16, 4),  blk, 0, stream>>>(xh, wk, khb, 4096, 256,  1024, 256);
  gemm_f16_kernel<2><<<dim3(16, 16), blk, 0, stream>>>(xh, wv, vt,  4096, 1024, 1024, 4096);
  // Column softmax stats, then fused attn*V, then output projection (fp32 out).
  colstats_kernel<<<512, blk, 0, stream>>>(qsw, khb, mxs, css);
  attn_o_kernel<<<512, blk, 0, stream>>>(qsw, khb, vt, mxs, css, oh);
  gemm_f16_kernel<3><<<dim3(16, 16), blk, 0, stream>>>(oh, wo, d_out, 4096, 1024, 1024, 1024);
}
